// Lfm2MoeSparseMoeBlock_32736240730968
// MI455X (gfx1250) — matrix operand order repaired and checked by reading, not compiled, not run on hardware
//
#include <hip/hip_runtime.h>
#include <math.h>

// ---- problem constants (match reference) ----
constexpr int TOK   = 2048;  // B*S tokens
constexpr int HID   = 2048;  // hidden dim H
constexpr int NEXP  = 32;    // experts E
constexpr int INTER = 1024;  // intermediate I
constexpr int TOPK  = 4;
constexpr float NORM_EPS = 1e-6f;
constexpr float ROUTED_SCALING = 1.0f;

// ---- vector types ----
typedef __bf16 bf16_t;
typedef bf16_t  v16bf  __attribute__((ext_vector_type(16)));
typedef bf16_t  bf16x2 __attribute__((ext_vector_type(2)));
typedef float   v8f    __attribute__((ext_vector_type(8)));
typedef unsigned int u32x4 __attribute__((ext_vector_type(4)));
typedef int     i32x4  __attribute__((ext_vector_type(4)));
typedef int     i32x8  __attribute__((ext_vector_type(8)));

// ---- LDS layout for moe_kernel (dynamic shared) ----
constexpr int SMEM_X_BYTES = 16 * HID * 4;          // fp32 gathered X tile (128 KB)
constexpr int SMEM_G_OFF   = SMEM_X_BYTES;
constexpr int SMEM_G_BYTES = 16 * INTER * 2;        // bf16 g tile (32 KB)
constexpr int SMEM_TOK_OFF = SMEM_G_OFF + SMEM_G_BYTES;
constexpr int SMEM_WT_OFF  = SMEM_TOK_OFF + 64;
constexpr int SMEM_TOTAL   = SMEM_WT_OFF + 64;      // 163968 B < 320 KB/WGP
static_assert(SMEM_TOTAL <= 320 * 1024, "LDS budget");

// fast sigmoid denominator reciprocal: v_rcp_f32 (bf16-level accuracy is fine here)
__device__ inline float fast_rcp(float x) { return __builtin_amdgcn_rcpf(x); }

// ============================================================
// Kernel 1: zero output accumulator region + expert counters
// ============================================================
__global__ void moe_zero_kernel(float* __restrict__ out, int n, int* __restrict__ cnt) {
    int gid = blockIdx.x * blockDim.x + threadIdx.x;
    if (gid < n)    out[gid] = 0.0f;
    if (gid < NEXP) cnt[gid] = 0;
}

// ============================================================
// Kernel 2: router — one wave32 per token
// ============================================================
__global__ void __launch_bounds__(256) moe_router_kernel(
    const float* __restrict__ x, const float* __restrict__ gw,
    const float* __restrict__ bias, float* __restrict__ logits_out,
    int* __restrict__ cnt, int* __restrict__ tok, float* __restrict__ wts) {

    const int lane = threadIdx.x & 31;
    const int wave = threadIdx.x >> 5;
    const int t = blockIdx.x * (blockDim.x >> 5) + wave;
    if (t >= TOK) return;

    const float* xr = x + (size_t)t * HID;
    float acc[NEXP];
#pragma unroll
    for (int e = 0; e < NEXP; ++e) acc[e] = 0.f;

    // lanes split H; coalesced along h for every expert row of gate_w
    for (int h = lane; h < HID; h += 32) {
        float xv = xr[h];
#pragma unroll
        for (int e = 0; e < NEXP; ++e) acc[e] += xv * gw[(size_t)e * HID + h];
    }

    // reduce each expert's partial across the wave; lane e keeps logit_e
    float logit = 0.f;
#pragma unroll
    for (int e = 0; e < NEXP; ++e) {
        float r = acc[e];
#pragma unroll
        for (int off = 16; off >= 1; off >>= 1) r += __shfl_xor(r, off, 32);
        if (lane == e) logit = r;
    }

    logits_out[(size_t)t * NEXP + lane] = logit;

    const float sig = 1.f / (1.f + __expf(-logit));   // router: keep accurate path
    float score = sig + bias[lane];

    // iterative top-4 argmax (ties -> lowest index, matching lax.top_k)
    int sel[TOPK];
    float wsel[TOPK];
    float v = score;
#pragma unroll
    for (int k = 0; k < TOPK; ++k) {
        float bv = v; int bi = lane;
#pragma unroll
        for (int off = 16; off >= 1; off >>= 1) {
            float ov = __shfl_xor(bv, off, 32);
            int   oi = __shfl_xor(bi, off, 32);
            if (ov > bv || (ov == bv && oi < bi)) { bv = ov; bi = oi; }
        }
        sel[k]  = bi;
        wsel[k] = __shfl(sig, bi, 32);   // raw sigmoid from winning lane
        if (lane == bi) v = -__builtin_huge_valf();
    }

    if (lane == 0) {
        float s = wsel[0] + wsel[1] + wsel[2] + wsel[3] + NORM_EPS;
        float inv = ROUTED_SCALING / s;
#pragma unroll
        for (int k = 0; k < TOPK; ++k) {
            int e = sel[k];
            int pos = atomicAdd(&cnt[e], 1);
            tok[(size_t)e * TOK + pos] = t;
            wts[(size_t)e * TOK + pos] = wsel[k] * inv;
        }
    }
}

// ============================================================
// Kernel 3: expert SwiGLU + down-proj, WMMA bf16, TDM gather
// block = 256 threads (8 waves); grid = (E, tiles_of_16_tokens)
// ============================================================
__global__ void __launch_bounds__(256, 1) moe_expert_kernel(
    const float* __restrict__ x,
    const float* __restrict__ w1, const float* __restrict__ w3,
    const float* __restrict__ w2,
    const int* __restrict__ cnt, const int* __restrict__ tok,
    const float* __restrict__ wts, float* __restrict__ out) {

    extern __shared__ __align__(16) char smem[];
    float*  xt    = (float*)smem;                       // [16][HID] fp32
    bf16_t* gtile = (bf16_t*)(smem + SMEM_G_OFF);       // [16][INTER] bf16
    int*    s_tok = (int*)(smem + SMEM_TOK_OFF);        // [16]
    float*  s_wt  = (float*)(smem + SMEM_WT_OFF);       // [16]

    const int e    = blockIdx.x;
    const int tile = blockIdx.y;
    const int cntE = cnt[e];
    if (tile * 16 >= cntE) return;   // uniform per block

    const int tid = threadIdx.x;
    if (tid < 16) {
        int row = tile * 16 + tid;
        if (row < cntE) {
            s_tok[tid] = tok[(size_t)e * TOK + row];
            s_wt[tid]  = wts[(size_t)e * TOK + row];
        } else {
            s_tok[tid] = 0;          // gather a valid row, weight 0
            s_wt[tid]  = 0.f;
        }
    }
    __syncthreads();

    // ---- stage gathered token rows into LDS ----
#if defined(__has_builtin) && __has_builtin(__builtin_amdgcn_tensor_load_to_lds) && __has_builtin(__builtin_amdgcn_s_wait_tensorcnt)
    // TDM gather mode: 16 x 16-bit row indices, 16 rows of HID fp32 -> LDS
    if (tid < 32) {                  // single wave issues the DMA
        unsigned ldsoff = (unsigned)(unsigned long long)(void*)xt;   // LDS aperture: addr[31:0]
        unsigned long long ga = (unsigned long long)(const void*)x;
        u32x4 g0;
        g0[0] = 0x80000001u;                              // count=1, gather_mode=1, 16-bit idx
        g0[1] = ldsoff;                                   // lds_addr
        g0[2] = (unsigned)(ga & 0xFFFFFFFFull);           // global_addr[31:0]
        g0[3] = (unsigned)((ga >> 32) & 0x1FFFFFFull) | (2u << 30);  // addr[56:32], type=2
        i32x8 g1;
        g1[0] = (int)(2u << 16);                          // data_size=4B; wg_mask=0
        g1[1] = (int)((HID & 0xFFFF) << 16);              // tensor_dim0[15:0]
        g1[2] = (int)(((HID >> 16) & 0xFFFF) | ((TOK & 0xFFFF) << 16)); // td0 hi | td1 lo
        g1[3] = (int)(((TOK >> 16) & 0xFFFF) | ((HID & 0xFFFF) << 16)); // td1 hi | tile_dim0=HID
        g1[4] = 16;                                       // tile_dim1 = #valid indices
        g1[5] = HID;                                      // tensor_dim0_stride[31:0]
        g1[6] = 0;
        g1[7] = 0;
        i32x4 g2, g3;
#pragma unroll
        for (int i = 0; i < 4; ++i) {
            g2[i] = (s_tok[2 * i]     & 0xFFFF) | ((s_tok[2 * i + 1] & 0xFFFF) << 16);
            g3[i] = (s_tok[8 + 2 * i] & 0xFFFF) | ((s_tok[9 + 2 * i] & 0xFFFF) << 16);
        }
        i32x8 g4 = {0, 0, 0, 0, 0, 0, 0, 0};              // unused trailing group (6-arg form)
        __builtin_amdgcn_tensor_load_to_lds(g0, g1, g2, g3, g4, 0);
        __builtin_amdgcn_s_wait_tensorcnt(0);
    }
#else
    for (int idx = tid; idx < 16 * HID; idx += 256) {
        int m = idx / HID, h = idx & (HID - 1);
        xt[idx] = x[(size_t)s_tok[m] * HID + h];
    }
#endif
    __syncthreads();

    const int lane = tid & 31;
    const int wave = tid >> 5;
    const int half = lane >> 4;      // K half-wave offset per WMMA layout
    const int ncol = lane & 15;      // N / B-matrix column, C/D column
    const int mrow = lane & 15;      // A-matrix row

    // ---------------- Phase A: h1/h3 GEMMs + SwiGLU -> g tile ----------------
    const size_t wbase = (size_t)e * HID * INTER;
    const float* w1e = w1 + wbase;
    const float* w3e = w3 + wbase;
    const float* arow = xt + (size_t)mrow * HID;

    for (int nt = wave; nt < INTER / 16; nt += 8) {
        const int n0 = nt * 16;
        v8f acc1 = {}; v8f acc3 = {};
        const float* b1p = w1e + (size_t)(half * 16) * INTER + (n0 + ncol);
        const float* b3p = w3e + (size_t)(half * 16) * INTER + (n0 + ncol);
#pragma unroll 1
        for (int ks = 0; ks < HID / 32; ++ks) {
            const int kb = ks * 32;
            // A fragment: 16x32 bf16, converted from fp32 LDS
            v16bf a;
#pragma unroll
            for (int v = 0; v < 8; ++v) {
                int k = kb + 2 * v + half * 16;
                float2 f = *(const float2*)(arow + k);
                a[2 * v]     = (bf16_t)f.x;
                a[2 * v + 1] = (bf16_t)f.y;
            }
            // B fragments: 32x16 slices of w1/w3 (row stride INTER)
            const float* c1 = b1p + (size_t)kb * INTER;
            const float* c3 = b3p + (size_t)kb * INTER;
            __builtin_prefetch(c1 + (size_t)32 * INTER, 0, 0);
            v16bf b1, b3;
#pragma unroll
            for (int v = 0; v < 8; ++v) {
                b1[2 * v]     = (bf16_t)c1[(size_t)(2 * v) * INTER];
                b1[2 * v + 1] = (bf16_t)c1[(size_t)(2 * v + 1) * INTER];
                b3[2 * v]     = (bf16_t)c3[(size_t)(2 * v) * INTER];
                b3[2 * v + 1] = (bf16_t)c3[(size_t)(2 * v + 1) * INTER];
            }
            acc1 = __builtin_amdgcn_wmma_f32_16x16x32_bf16(false, a, false, b1, (short)0, acc1, false, false);
            acc3 = __builtin_amdgcn_wmma_f32_16x16x32_bf16(false, a, false, b3, (short)0, acc3, false, false);
        }
        // SwiGLU (silu via v_exp + v_rcp, no IEEE-div expansion) and store bf16 g tile
#pragma unroll
        for (int r = 0; r < 8; ++r) {
            float h1 = acc1[r], h3 = acc3[r];
            float gv = h1 * fast_rcp(1.f + __expf(-h1)) * h3;
            gtile[(size_t)(r + 8 * half) * INTER + n0 + ncol] = (bf16_t)gv;
        }
    }
    __syncthreads();

    // ---------------- Phase B: out += weight * (g @ w2) ----------------
    const float* w2e = w2 + (size_t)e * INTER * HID;
    const bf16_t* grow = gtile + (size_t)mrow * INTER;
    float wrow[8]; int trow[8];
#pragma unroll
    for (int r = 0; r < 8; ++r) {
        wrow[r] = s_wt[r + 8 * half];
        trow[r] = s_tok[r + 8 * half];
    }

    for (int nt = wave; nt < HID / 16; nt += 8) {
        const int n0 = nt * 16;
        v8f acc = {};
        const float* bp = w2e + (size_t)(half * 16) * HID + (n0 + ncol);
#pragma unroll 1
        for (int ks = 0; ks < INTER / 32; ++ks) {
            const int kb = ks * 32;
            v16bf a;
#pragma unroll
            for (int v = 0; v < 8; ++v) {
                int k = kb + 2 * v + half * 16;
                bf16x2 p = *(const bf16x2*)(grow + k);
                a[2 * v]     = p.x;
                a[2 * v + 1] = p.y;
            }
            const float* c = bp + (size_t)kb * HID;
            __builtin_prefetch(c + (size_t)32 * HID, 0, 0);
            v16bf b;
#pragma unroll
            for (int v = 0; v < 8; ++v) {
                b[2 * v]     = (bf16_t)c[(size_t)(2 * v) * HID];
                b[2 * v + 1] = (bf16_t)c[(size_t)(2 * v + 1) * HID];
            }
            acc = __builtin_amdgcn_wmma_f32_16x16x32_bf16(false, a, false, b, (short)0, acc, false, false);
        }
#pragma unroll
        for (int r = 0; r < 8; ++r) {
            atomicAdd(&out[(size_t)trow[r] * HID + n0 + ncol], wrow[r] * acc[r]);
        }
    }
}

// ============================================================
// host launcher
// ============================================================
extern "C" void kernel_launch(void* const* d_in, const int* in_sizes, int n_in,
                              void* d_out, int out_size, void* d_ws, size_t ws_size,
                              hipStream_t stream) {
    (void)in_sizes; (void)n_in; (void)out_size; (void)ws_size;
    const float* x    = (const float*)d_in[0];
    const float* gw   = (const float*)d_in[1];
    const float* w1   = (const float*)d_in[2];
    const float* w3   = (const float*)d_in[3];
    const float* w2   = (const float*)d_in[4];
    const float* bias = (const float*)d_in[5];

    float* out    = (float*)d_out;                     // [TOK, HID]
    float* logits = out + (size_t)TOK * HID;           // [TOK, NEXP]

    // workspace: counters + per-expert token/weight lists
    int*   cnt = (int*)d_ws;
    int*   tok = (int*)((char*)d_ws + 256);
    float* wts = (float*)((char*)d_ws + 256 + sizeof(int) * (size_t)NEXP * TOK);

    moe_zero_kernel<<<(TOK * HID + 255) / 256, 256, 0, stream>>>(out, TOK * HID, cnt);

    moe_router_kernel<<<TOK / 8, 256, 0, stream>>>(x, gw, bias, logits, cnt, tok, wts);

    (void)hipFuncSetAttribute(reinterpret_cast<const void*>(moe_expert_kernel),
                              hipFuncAttributeMaxDynamicSharedMemorySize, SMEM_TOTAL);
    moe_expert_kernel<<<dim3(NEXP, (TOK + 15) / 16), 256, SMEM_TOTAL, stream>>>(
        x, w1, w3, w2, cnt, tok, wts, out);
}
